// GraphAttentionLayer_62096637166286
// MI455X (gfx1250) — compile-verified, matrix operand order checked by reading, not run on hardware
//
#include <hip/hip_runtime.h>
#include <hip/hip_bf16.h>
#include <stdint.h>

typedef __attribute__((ext_vector_type(16))) __bf16 v16bf;
typedef __attribute__((ext_vector_type(8)))  __bf16 v8bf;
typedef __attribute__((ext_vector_type(8)))  float  v8f;
typedef __attribute__((ext_vector_type(4)))  float  v4f;
typedef __attribute__((ext_vector_type(4)))  int    v4i;

#define GAT_ALPHA 0.2f
#define GAT_NEG_INF -9000000000000000.0f

static constexpr int GN = 8192;   // nodes
static constexpr int GF = 256;    // features

// ---------------------------------------------------------------------------
// Kernel 0a: f32 -> bf16 convert (input, W1, W2)
// ---------------------------------------------------------------------------
__global__ void cvt_bf16_kernel(const float* __restrict__ src,
                                unsigned short* __restrict__ dst, int n) {
    int i = blockIdx.x * blockDim.x + threadIdx.x;
    if (i < n) {
        __bf16 v = (__bf16)src[i];
        dst[i] = *(unsigned short*)&v;
    }
}

// ---------------------------------------------------------------------------
// Kernel 0b: tiled transpose  input[N][F] f32  ->  XT[F][N] bf16
// ---------------------------------------------------------------------------
__global__ void transpose_bf16_kernel(const float* __restrict__ src,
                                      unsigned short* __restrict__ dstT) {
    __shared__ float tile[32][33];
    int j0 = blockIdx.x * 32;           // node dim
    int n0 = blockIdx.y * 32;           // feature dim
    int tx = threadIdx.x, ty = threadIdx.y;   // block (32, 8)
    #pragma unroll
    for (int r = 0; r < 32; r += 8)
        tile[ty + r][tx] = src[(size_t)(j0 + ty + r) * GF + n0 + tx];
    __syncthreads();
    #pragma unroll
    for (int r = 0; r < 32; r += 8) {
        __bf16 v = (__bf16)tile[tx][ty + r];
        dstT[(size_t)(n0 + ty + r) * GN + j0 + tx] = *(unsigned short*)&v;
    }
}

// ---------------------------------------------------------------------------
// Helpers
// ---------------------------------------------------------------------------
__device__ __forceinline__ v16bf load_frag_row(const __bf16* __restrict__ row,
                                               int kbase, int hi) {
    // 16-bit A/B fragment: per lane two contiguous 8-element K-chunks:
    //   K = kbase + hi*8 + [0..7]  and  K = kbase + 16 + hi*8 + [0..7]
    v8bf lo = *(const v8bf*)(row + kbase + hi * 8);
    v8bf hh = *(const v8bf*)(row + kbase + 16 + hi * 8);
    return __builtin_shufflevector(lo, hh, 0,1,2,3,4,5,6,7,8,9,10,11,12,13,14,15);
}

__device__ __forceinline__ void async_load_b128(uint32_t lds_addr, const void* gptr) {
    // GLOBAL_LOAD_ASYNC_TO_LDS_B128, GV mode: per-lane 16B memory -> LDS copy,
    // tracked with ASYNCcnt.
    asm volatile("global_load_async_to_lds_b128 %0, %1, off"
                 :: "v"(lds_addr), "v"(gptr) : "memory");
}

__device__ __forceinline__ void wait_asynccnt0() {
    asm volatile("s_wait_asynccnt 0x0" ::: "memory");
}

// ---------------------------------------------------------------------------
// Kernel 1: s1[i] = tanh(X@W1^T + b1) @ u ; s2[i] = tanh(X@W2^T + b2) @ v
// One wave per 16-row strip, WMMA bf16 16x16x32, K=256, 16 N-tiles.
// ---------------------------------------------------------------------------
__global__ void __launch_bounds__(128)
scores_kernel(const unsigned short* __restrict__ Xbf_u,
              const unsigned short* __restrict__ W1bf_u,
              const unsigned short* __restrict__ W2bf_u,
              const float* __restrict__ b1, const float* __restrict__ b2,
              const float* __restrict__ aw,
              float* __restrict__ s1, float* __restrict__ s2) {
    const __bf16* Xbf  = (const __bf16*)Xbf_u;
    const __bf16* W1bf = (const __bf16*)W1bf_u;
    const __bf16* W2bf = (const __bf16*)W2bf_u;

    int wave = threadIdx.x >> 5;
    int lane = threadIdx.x & 31;
    int m  = lane & 15;      // row (A) / col (B) within tile
    int hi = lane >> 4;      // K-chunk selector
    int i0 = (blockIdx.x * 4 + wave) * 16;

    const __bf16* arow = Xbf + (size_t)(i0 + m) * GF;

    // Preload the 8 A fragments (rows reused across all 16 N-tiles)
    v16bf afr[8];
    #pragma unroll
    for (int kk = 0; kk < 8; ++kk)
        afr[kk] = load_frag_row(arow, kk * 32, hi);

    v8f p1 = {}, p2 = {};
    #pragma unroll 1
    for (int nt = 0; nt < 16; ++nt) {
        int n = nt * 16 + m;
        const __bf16* w1row = W1bf + (size_t)n * GF;
        const __bf16* w2row = W2bf + (size_t)n * GF;
        v8f acc1 = {}, acc2 = {};
        #pragma unroll
        for (int kk = 0; kk < 8; ++kk) {
            v16bf bf1 = load_frag_row(w1row, kk * 32, hi);
            v16bf bf2 = load_frag_row(w2row, kk * 32, hi);
            acc1 = __builtin_amdgcn_wmma_f32_16x16x32_bf16(
                false, afr[kk], false, bf1, (short)0, acc1, false, false);
            acc2 = __builtin_amdgcn_wmma_f32_16x16x32_bf16(
                false, afr[kk], false, bf2, (short)0, acc2, false, false);
        }
        float un  = aw[n];
        float vn  = aw[GF + n];
        float bb1 = b1[n];
        float bb2 = b2[n];
        // C layout: element r <-> row (r + hi*8), col n
        #pragma unroll
        for (int r = 0; r < 8; ++r) {
            p1[r] += tanhf(acc1[r] + bb1) * un;
            p2[r] += tanhf(acc2[r] + bb2) * vn;
        }
    }
    // Reduce across the 16 lanes of each half (xor masks stay within half)
    #pragma unroll
    for (int r = 0; r < 8; ++r) {
        float a = p1[r], b = p2[r];
        #pragma unroll
        for (int d = 1; d < 16; d <<= 1) {
            a += __shfl_xor(a, d);
            b += __shfl_xor(b, d);
        }
        p1[r] = a; p2[r] = b;
    }
    if (m == 0) {
        #pragma unroll
        for (int r = 0; r < 8; ++r) {
            s1[i0 + hi * 8 + r] = p1[r];
            s2[i0 + hi * 8 + r] = p2[r];
        }
    }
}

// ---------------------------------------------------------------------------
// Kernel 2: fused mask + softmax + attention write + h' = att @ X
//
// Block = 4 waves / 64 rows; each wave owns a unique 16-row tile.
// Pass 1 (per-wave, independent): stream adj once, accumulate row sums,
//          pack adjacency bits into an LDS mask (16 KB / wave).
// Pass 2 (lockstep): per 32-column step, stage the 32x256 bf16 B-slab of X^T
//          into LDS with async-to-LDS copies (double buffered, ASYNCcnt),
//          recompute attention from the LDS mask, write normalized attention,
//          and WMMA-accumulate h'.
// ---------------------------------------------------------------------------
__device__ __forceinline__ float att_p(float s1b, float z, int av) {
    float e = s1b + z;
    e = e > 0.f ? e : (GAT_ALPHA * e);
    return __expf(av > 0 ? e : GAT_NEG_INF);   // expf(-9e15) underflows to 0
}

static constexpr int BROW = 80;                  // padded LDS row (64B data + 16B pad)
static constexpr int BBUF = 256 * BROW;          // 20480 B per B-slab buffer

__global__ void __launch_bounds__(128)
attn_kernel(const int* __restrict__ adj,
            const float* __restrict__ s1, const float* __restrict__ s2,
            const float* __restrict__ ab,
            const unsigned short* __restrict__ XbfT_u,
            float* __restrict__ att, float* __restrict__ hprime) {
    __shared__ __align__(16) unsigned char lds_B[2 * BBUF];   // 40 KB
    __shared__ unsigned int lds_mask[4 * 4096];               // 64 KB: [wave][t][m]

    int tid  = threadIdx.x;
    int wave = tid >> 5;
    int lane = tid & 31;
    int m    = lane & 15;
    int hi   = lane >> 4;
    int i0w  = blockIdx.x * 64 + wave * 16;
    int row  = i0w + m;

    const float s1i = s1[row] + ab[0];
    const int* adjrow = adj + (size_t)row * GN;
    const unsigned maskbase = wave * 4096;

    // ---- pass 1: row sums + adjacency bit-mask to LDS --------------------
    float psum = 0.f;
    const int b0 = hi * 8;
    for (int t = 0; t < GN / 32; ++t) {
        int c0 = t * 32 + b0;
        const int* ap = adjrow + c0;
        __builtin_prefetch(ap + 32, 0, 0);          // next 32-column tile
        v4i a0 = *(const v4i*)(ap);
        v4i a1 = *(const v4i*)(ap + 4);
        v4i a2 = *(const v4i*)(ap + 16);
        v4i a3 = *(const v4i*)(ap + 20);
        v4f z0 = *(const v4f*)(s2 + c0);
        v4f z1 = *(const v4f*)(s2 + c0 + 4);
        v4f z2 = *(const v4f*)(s2 + c0 + 16);
        v4f z3 = *(const v4f*)(s2 + c0 + 20);
        unsigned bits = 0;
        #pragma unroll
        for (int q = 0; q < 4; ++q) {
            bits |= (unsigned)(a0[q] > 0) << (b0 + q);
            bits |= (unsigned)(a1[q] > 0) << (b0 + 4 + q);
            bits |= (unsigned)(a2[q] > 0) << (b0 + 16 + q);
            bits |= (unsigned)(a3[q] > 0) << (b0 + 20 + q);
            psum += att_p(s1i, z0[q], a0[q]);
            psum += att_p(s1i, z1[q], a1[q]);
            psum += att_p(s1i, z2[q], a2[q]);
            psum += att_p(s1i, z3[q], a3[q]);
        }
        unsigned word = bits | (unsigned)__shfl_xor((int)bits, 16);
        if (hi == 0) lds_mask[maskbase + t * 16 + m] = word;
    }
    const float inv = 1.0f / (psum + __shfl_xor(psum, 16));

    // ---- pass 2: lockstep over column tiles, B staged in LDS -------------
    const uint32_t ldsB0 = (uint32_t)(uintptr_t)(&lds_B[0]);
    const char* Xb = (const char*)XbfT_u;

    // Stage B-slab for tile t into buffer buf: 256 rows x 64B, 80B LDS pitch.
    auto issueB = [&](int t, int buf) {
        const char* gbase = Xb + (size_t)t * 64;
        #pragma unroll
        for (int it = 0; it < 8; ++it) {
            int c = tid + it * 128;               // 1024 16B chunks
            int n = c >> 2;
            int part = (c & 3) * 16;
            async_load_b128(ldsB0 + buf * BBUF + n * BROW + part,
                            gbase + (size_t)n * (GN * 2) + part);
        }
    };

    issueB(0, 0);
    v8f acc[16] = {};
    for (int t = 0; t < GN / 32; ++t) {
        int cur = t & 1;
        wait_asynccnt0();                         // my copies into buf[cur] done
        __syncthreads();                          // everyone's copies done
        if (t + 1 < GN / 32) issueB(t + 1, 1 - cur);

        unsigned word = lds_mask[maskbase + t * 16 + m];
        int c0 = t * 32 + b0;
        v4f z0 = *(const v4f*)(s2 + c0);
        v4f z1 = *(const v4f*)(s2 + c0 + 4);
        v4f z2 = *(const v4f*)(s2 + c0 + 16);
        v4f z3 = *(const v4f*)(s2 + c0 + 20);

        float p[16];
        #pragma unroll
        for (int q = 0; q < 4; ++q) {
            p[q]      = att_p(s1i, z0[q], (int)((word >> (b0 + q)) & 1u)) * inv;
            p[4 + q]  = att_p(s1i, z1[q], (int)((word >> (b0 + 4 + q)) & 1u)) * inv;
            p[8 + q]  = att_p(s1i, z2[q], (int)((word >> (b0 + 16 + q)) & 1u)) * inv;
            p[12 + q] = att_p(s1i, z3[q], (int)((word >> (b0 + 20 + q)) & 1u)) * inv;
        }

        // Write attention row chunks (f32, b128 stores)
        float* aout = att + (size_t)row * GN + c0;
        v4f w0 = {p[0], p[1], p[2], p[3]};
        v4f w1 = {p[4], p[5], p[6], p[7]};
        v4f w2 = {p[8], p[9], p[10], p[11]};
        v4f w3 = {p[12], p[13], p[14], p[15]};
        *(v4f*)(aout)      = w0;
        *(v4f*)(aout + 4)  = w1;
        *(v4f*)(aout + 16) = w2;
        *(v4f*)(aout + 20) = w3;

        // A fragment (att tile 16x32 bf16): p[] is already in fragment order.
        v16bf afrag;
        #pragma unroll
        for (int j = 0; j < 16; ++j) afrag[j] = (__bf16)p[j];

        const unsigned char* bbase = lds_B + cur * BBUF;
        #pragma unroll
        for (int nt = 0; nt < 16; ++nt) {
            int n = nt * 16 + m;
            v8bf lo = *(const v8bf*)(bbase + n * BROW + hi * 16);
            v8bf hh = *(const v8bf*)(bbase + n * BROW + 32 + hi * 16);
            v16bf bfrag = __builtin_shufflevector(
                lo, hh, 0,1,2,3,4,5,6,7,8,9,10,11,12,13,14,15);
            acc[nt] = __builtin_amdgcn_wmma_f32_16x16x32_bf16(
                false, afrag, false, bfrag, (short)0, acc[nt], false, false);
        }
    }

    // ---- write h' (each wave owns its complete 16x256 tile) --------------
    // C layout: acc[nt][r] <-> row (i0w + r + hi*8), col (nt*16 + m)
    #pragma unroll
    for (int nt = 0; nt < 16; ++nt) {
        #pragma unroll
        for (int r = 0; r < 8; ++r)
            hprime[(size_t)(i0w + r + hi * 8) * GF + nt * 16 + m] = acc[nt][r];
    }
}

// ---------------------------------------------------------------------------
// Host launcher
// ---------------------------------------------------------------------------
extern "C" void kernel_launch(void* const* d_in, const int* in_sizes, int n_in,
                              void* d_out, int out_size, void* d_ws, size_t ws_size,
                              hipStream_t stream) {
    const float* input = (const float*)d_in[0];
    const int*   adj   = (const int*)d_in[1];
    const float* W1w   = (const float*)d_in[2];
    const float* W1b   = (const float*)d_in[3];
    const float* W2w   = (const float*)d_in[4];
    const float* W2b   = (const float*)d_in[5];
    const float* aw    = (const float*)d_in[6];
    const float* ab    = (const float*)d_in[7];

    char* ws = (char*)d_ws;
    unsigned short* Xbf  = (unsigned short*)(ws);                                // 4 MB
    unsigned short* XbfT = (unsigned short*)(ws + ((size_t)4 << 20));            // 4 MB
    unsigned short* W1bf = (unsigned short*)(ws + ((size_t)8 << 20));            // 128 KB
    unsigned short* W2bf = (unsigned short*)(ws + ((size_t)8 << 20) + (128 << 10));
    float*          s1   = (float*)(ws + ((size_t)8 << 20) + (256 << 10));       // 32 KB
    float*          s2   = (float*)(ws + ((size_t)8 << 20) + (288 << 10));       // 32 KB

    float* hprime = (float*)d_out;                       // 8192*256
    float* att    = (float*)d_out + (size_t)GN * GF;     // 8192*8192

    cvt_bf16_kernel<<<(GN * GF + 255) / 256, 256, 0, stream>>>(input, Xbf, GN * GF);
    cvt_bf16_kernel<<<(GF * GF + 255) / 256, 256, 0, stream>>>(W1w, W1bf, GF * GF);
    cvt_bf16_kernel<<<(GF * GF + 255) / 256, 256, 0, stream>>>(W2w, W2bf, GF * GF);
    transpose_bf16_kernel<<<dim3(GN / 32, GF / 32), dim3(32, 8), 0, stream>>>(input, XbfT);

    scores_kernel<<<GN / 64, 128, 0, stream>>>(Xbf, W1bf, W2bf, W1b, W2b, aw, s1, s2);
    attn_kernel<<<GN / 64, 128, 0, stream>>>(adj, s1, s2, ab, XbfT, att, hprime);
}
